// GenODE_55173149884435
// MI455X (gfx1250) — matrix operand order repaired and checked by reading, not compiled, not run on hardware
//
#include <hip/hip_runtime.h>
#include <hip/hip_bf16.h>

typedef __attribute__((ext_vector_type(16))) _Float16 v16h;
typedef __attribute__((ext_vector_type(8)))  _Float16 v8h;
typedef __attribute__((ext_vector_type(8)))  float    v8f;

#define EPSF 1e-5f

// ---------------------------------------------------------------------------
// fp32 -> f16 weight conversion (one-time per launch; weights stay resident)
// ---------------------------------------------------------------------------
__global__ __launch_bounds__(256) void f32_to_f16_k(const float* __restrict__ src,
                                                    _Float16* __restrict__ dst, int n) {
    int i = blockIdx.x * 256 + threadIdx.x;
    if (i < n) dst[i] = (_Float16)src[i];
}

// ---------------------------------------------------------------------------
// WMMA GEMM:  C[M,N] = act( A[M,K] @ W[N,K]^T + bias[N] )
//
// Software-pipelined, double-buffered LDS. Two tile configs (8 wave32s):
//   BM=64, BN=128 : wave tile 16x64, 4 accumulators, A-frag reused 4x  (big M)
//   BM=32, BN= 64 : wave tile 16x16, 1 accumulator                    (small M)
// Per K-iteration: issue next tile's global_load_b128s first, run WMMAs from
// the current LDS buffer, then convert+store into the other buffer; a single
// barrier per iteration. A fp32->f16 converted in flight; W pre-converted f16.
// FULL=true  : N multiple of 64 -> zero guards.  ACT: 0 none, 1 PReLU.
// ---------------------------------------------------------------------------
template <int ACT, int BM, int BN, bool FULL>
__global__ __launch_bounds__(256) void gemm_tile(const float*    __restrict__ A,
                                                 const _Float16* __restrict__ W,
                                                 const float*    __restrict__ bias,
                                                 float*          __restrict__ C,
                                                 int M, int N, int K,
                                                 const float*    __restrict__ alpha_p) {
    constexpr int BK   = 32;
    constexpr int NSUB = (BM == 64) ? 4 : 1;        // 16-wide N sub-tiles per wave
    constexpr int MW   = BM / 16;                   // M strips (waves along M)
    constexpr int SB   = (BN * 4) / 256;            // W segments per thread

    __shared__ _Float16 As[2][BM][BK + 8];          // stride 80B: keeps 16B align
    __shared__ _Float16 Bs[2][BN][BK + 8];

    const int tid  = threadIdx.x;
    const int wave = tid >> 5;
    const int lane = tid & 31;
    const int wm   = wave % MW;
    const int wn   = wave / MW;
    const int bm   = blockIdx.y * BM;
    const int bn   = blockIdx.x * BN;
    const int sel  = lane >> 4;
    const int mr   = lane & 15;

    // staging coordinates: one aligned 8-half segment per thread per tile
    const int  ar   = tid >> 2, ac = (tid & 3) * 8;
    const bool aact = (BM * 4 >= 256) || (tid < BM * 4);

    v8f acc[NSUB];
#pragma unroll
    for (int s = 0; s < NSUB; ++s)
#pragma unroll
        for (int r = 0; r < 8; ++r) acc[s][r] = 0.f;

    // ---- prologue: stage K-tile 0 into buffer 0 ----
    if (aact) {
        const float4* ap =
            reinterpret_cast<const float4*>(&A[(size_t)(bm + ar) * K + ac]);
        float4 f0 = ap[0], f1 = ap[1];
        v8h h;
        h[0] = (_Float16)f0.x; h[1] = (_Float16)f0.y;
        h[2] = (_Float16)f0.z; h[3] = (_Float16)f0.w;
        h[4] = (_Float16)f1.x; h[5] = (_Float16)f1.y;
        h[6] = (_Float16)f1.z; h[7] = (_Float16)f1.w;
        *reinterpret_cast<v8h*>(&As[0][ar][ac]) = h;
    }
#pragma unroll
    for (int i = 0; i < SB; ++i) {
        int seg = tid + i * 256;
        int r = seg >> 2, c = (seg & 3) * 8;
        v8h h = {};
        if (FULL || (bn + r) < N)
            h = *reinterpret_cast<const v8h*>(&W[(size_t)(bn + r) * K + c]);
        *reinterpret_cast<v8h*>(&Bs[0][r][c]) = h;
    }
    __syncthreads();

    const int nt = K / BK;
    for (int it = 0; it < nt; ++it) {
        const int  cur  = it & 1;
        const int  nxt  = cur ^ 1;
        const bool more = (it + 1) < nt;
        const int  kn   = (it + 1) * BK;

        // ---- issue next tile's global loads before computing ----
        float4 f0 = {}, f1 = {};
        v8h    wreg[SB] = {};
        if (more) {
            if (aact) {
                const float4* ap = reinterpret_cast<const float4*>(
                    &A[(size_t)(bm + ar) * K + kn + ac]);
                f0 = ap[0];
                f1 = ap[1];
            }
#pragma unroll
            for (int i = 0; i < SB; ++i) {
                int seg = tid + i * 256;
                int r = seg >> 2, c = (seg & 3) * 8;
                if (FULL || (bn + r) < N)
                    wreg[i] = *reinterpret_cast<const v8h*>(
                        &W[(size_t)(bn + r) * K + kn + c]);
            }
        }

        // ---- compute from current buffer ----
        // A fragment (16-bit A 16x32), same k order as B:
        // lanes 0-15: K 0..15 ; lanes 16-31: K 16..31
        v16h a;
#pragma unroll
        for (int j = 0; j < 8; ++j) {
            a[j]     = As[cur][wm * 16 + mr][sel * 16 + j];
            a[j + 8] = As[cur][wm * 16 + mr][sel * 16 + 8 + j];
        }
        // B fragments (16-bit B 32x16): lanes 0-15 K=0..15, lanes 16-31 K=16..31
        v16h bf[NSUB];
#pragma unroll
        for (int s = 0; s < NSUB; ++s)
#pragma unroll
            for (int j = 0; j < 16; ++j)
                bf[s][j] = Bs[cur][wn * (NSUB * 16) + s * 16 + mr][sel * 16 + j];
#pragma unroll
        for (int s = 0; s < NSUB; ++s)
            acc[s] = __builtin_amdgcn_wmma_f32_16x16x32_f16(false, a, false, bf[s],
                                                            (short)0, acc[s],
                                                            false, false);

        // ---- store next tile into the other buffer ----
        if (more) {
            if (aact) {
                v8h h;
                h[0] = (_Float16)f0.x; h[1] = (_Float16)f0.y;
                h[2] = (_Float16)f0.z; h[3] = (_Float16)f0.w;
                h[4] = (_Float16)f1.x; h[5] = (_Float16)f1.y;
                h[6] = (_Float16)f1.z; h[7] = (_Float16)f1.w;
                *reinterpret_cast<v8h*>(&As[nxt][ar][ac]) = h;
            }
#pragma unroll
            for (int i = 0; i < SB; ++i) {
                int seg = tid + i * 256;
                int r = seg >> 2, c = (seg & 3) * 8;
                *reinterpret_cast<v8h*>(&Bs[nxt][r][c]) = wreg[i];
            }
        }
        __syncthreads();
    }

    // Epilogue: C layout — col = lane&15, row = vgpr + (lane>=16 ? 8 : 0)
    const float alpha = (ACT == 1) ? *alpha_p : 0.f;
    const int rbase = bm + wm * 16 + sel * 8;
#pragma unroll
    for (int s = 0; s < NSUB; ++s) {
        const int col = bn + wn * (NSUB * 16) + s * 16 + mr;
        if (FULL || col < N) {
            const float bv = bias[col];
#pragma unroll
            for (int r = 0; r < 8; ++r) {
                float v = acc[s][r] + bv;
                if (ACT == 1) v = (v >= 0.f) ? v : alpha * v;
                C[(size_t)(rbase + r) * N + col] = v;
            }
        }
    }
}

// ---------------------------------------------------------------------------
// Row-wise LayerNorm (biased var) + ELU.  One workgroup per row, N features.
// ---------------------------------------------------------------------------
__global__ __launch_bounds__(256) void layernorm_elu_k(const float* __restrict__ X,
                                                       const float* __restrict__ g,
                                                       const float* __restrict__ be,
                                                       float* __restrict__ Y, int N) {
    const int row = blockIdx.x;
    const float* x = X + (size_t)row * N;
    float s = 0.f, ss = 0.f;
    for (int j = threadIdx.x; j < N; j += 256) {
        float v = x[j];
        s += v; ss += v * v;
    }
    __shared__ float rs[256], rq[256];
    rs[threadIdx.x] = s; rq[threadIdx.x] = ss;
    __syncthreads();
    for (int off = 128; off > 0; off >>= 1) {
        if (threadIdx.x < off) {
            rs[threadIdx.x] += rs[threadIdx.x + off];
            rq[threadIdx.x] += rq[threadIdx.x + off];
        }
        __syncthreads();
    }
    const float mean = rs[0] / (float)N;
    const float var  = rq[0] / (float)N - mean * mean;
    const float inv  = rsqrtf(var + EPSF);
    for (int j = threadIdx.x; j < N; j += 256) {
        float v = (x[j] - mean) * inv * g[j] + be[j];
        Y[(size_t)row * N + j] = (v > 0.f) ? v : (__expf(v) - 1.f);  // ELU
    }
}

// ---------------------------------------------------------------------------
// BatchNorm1d, training mode (batch statistics, biased variance).
// One workgroup per feature; blockDim.x == B.
// ---------------------------------------------------------------------------
__global__ __launch_bounds__(256) void batchnorm_k(const float* __restrict__ H,
                                                   const float* __restrict__ gamma,
                                                   const float* __restrict__ beta,
                                                   float* __restrict__ Y, int B, int F) {
    const int f = blockIdx.x;
    const int t = threadIdx.x;
    const float v = H[(size_t)t * F + f];
    __shared__ float rs[256], rq[256];
    rs[t] = v; rq[t] = v * v;
    __syncthreads();
    for (int off = 128; off > 0; off >>= 1) {
        if (t < off) { rs[t] += rs[t + off]; rq[t] += rq[t + off]; }
        __syncthreads();
    }
    const float mean = rs[0] / (float)B;
    const float var  = rq[0] / (float)B - mean * mean;
    Y[(size_t)t * F + f] = (v - mean) * rsqrtf(var + EPSF) * gamma[f] + beta[f];
}

// ---------------------------------------------------------------------------
// Elementwise helpers for RK4
// ---------------------------------------------------------------------------
__global__ __launch_bounds__(256) void axpy_k(const float* __restrict__ y,
                                              const float* __restrict__ k,
                                              float c, float* __restrict__ out, int n) {
    int i = blockIdx.x * 256 + threadIdx.x;
    if (i < n) out[i] = y[i] + c * k[i];
}

__global__ __launch_bounds__(256) void rk4_k(float* __restrict__ y,
                                             const float* __restrict__ k1,
                                             const float* __restrict__ k2,
                                             const float* __restrict__ k3,
                                             const float* __restrict__ k4,
                                             float h, int n) {
    int i = blockIdx.x * 256 + threadIdx.x;
    if (i < n)
        y[i] += (h * (1.f / 6.f)) * (k1[i] + 2.f * k2[i] + 2.f * k3[i] + k4[i]);
}

__global__ __launch_bounds__(256) void scatter_sol_k(const float* __restrict__ y,
                                                     float* __restrict__ sol,
                                                     int ti, int B, int T, int L) {
    int i = blockIdx.x * 256 + threadIdx.x;
    if (i < B * L) {
        int b = i / L, l = i - b * L;
        sol[((size_t)b * T + ti) * L + l] = y[i];
    }
}

// ---------------------------------------------------------------------------
// Host orchestration
// ---------------------------------------------------------------------------
extern "C" void kernel_launch(void* const* d_in, const int* in_sizes, int n_in,
                              void* d_out, int out_size, void* d_ws, size_t ws_size,
                              hipStream_t stream) {
    (void)in_sizes; (void)n_in; (void)out_size; (void)ws_size;
    const int B = 256, IN = 128, L = 256, U = 1024, T = 64;

    // ---- input pointers (setup_inputs dict order, lists flattened) ----
    const float* x = (const float*)d_in[0];
    // d_in[1] = t (linspace 0..1, 64 pts) -> dt = 1/63 hardcoded
    const float *augWs[7], *augBs[7], *augAl[6];
    for (int i = 0; i < 7; ++i) augWs[i] = (const float*)d_in[2 + i];
    for (int i = 0; i < 7; ++i) augBs[i] = (const float*)d_in[9 + i];
    for (int i = 0; i < 6; ++i) augAl[i] = (const float*)d_in[16 + i];
    const float* bn_g = (const float*)d_in[22];
    const float* bn_b = (const float*)d_in[23];
    const float *odeWs[5], *odeBs[5], *odeGs[4], *odeBes[4];
    for (int i = 0; i < 5; ++i) odeWs[i]  = (const float*)d_in[24 + i];
    for (int i = 0; i < 5; ++i) odeBs[i]  = (const float*)d_in[29 + i];
    for (int i = 0; i < 4; ++i) odeGs[i]  = (const float*)d_in[34 + i];
    for (int i = 0; i < 4; ++i) odeBes[i] = (const float*)d_in[38 + i];
    const float *decWs[7], *decBs[7], *decAl[6];
    for (int i = 0; i < 7; ++i) decWs[i] = (const float*)d_in[42 + i];
    for (int i = 0; i < 7; ++i) decBs[i] = (const float*)d_in[49 + i];
    for (int i = 0; i < 6; ++i) decAl[i] = (const float*)d_in[56 + i];

    const int augIn[7]  = {IN, U, U, U, U, U, U};
    const int augOut[7] = {U,  U, U, U, U, U, L};
    const int odeIn[5]  = {L,  U, U, U, U};
    const int odeOut[5] = {U,  U, U, U, L};
    const int decIn[7]  = {L,  U, U, U, U, U, U};
    const int decOut[7] = {U,  U, U, U, U, U, 1};

    // ---- workspace carve-out ----
    size_t off = 0;
    auto alloc = [&](size_t bytes) -> void* {
        off = (off + 255) & ~(size_t)255;
        void* p = (char*)d_ws + off;
        off += bytes;
        return p;
    };
    _Float16 *whAug[7], *whOde[5], *whDec[7];
    for (int i = 0; i < 7; ++i) whAug[i] = (_Float16*)alloc(sizeof(_Float16) * augOut[i] * augIn[i]);
    for (int i = 0; i < 5; ++i) whOde[i] = (_Float16*)alloc(sizeof(_Float16) * odeOut[i] * odeIn[i]);
    for (int i = 0; i < 7; ++i) whDec[i] = (_Float16*)alloc(sizeof(_Float16) * decOut[i] * decIn[i]);
    float* actA = (float*)alloc(sizeof(float) * B * U);
    float* actB = (float*)alloc(sizeof(float) * B * U);
    float* hbuf = (float*)alloc(sizeof(float) * B * L);
    float* y    = (float*)alloc(sizeof(float) * B * L);
    float* k1   = (float*)alloc(sizeof(float) * B * L);
    float* k2   = (float*)alloc(sizeof(float) * B * L);
    float* k3   = (float*)alloc(sizeof(float) * B * L);
    float* k4   = (float*)alloc(sizeof(float) * B * L);
    float* ytmp = (float*)alloc(sizeof(float) * B * L);
    float* sol  = (float*)alloc(sizeof(float) * (size_t)B * T * L);
    const int CH = 2048;
    float* dbufA = (float*)alloc(sizeof(float) * (size_t)CH * U);
    float* dbufB = (float*)alloc(sizeof(float) * (size_t)CH * U);

    // ---- convert weights to f16 ----
    auto conv = [&](const float* s, _Float16* d, int n) {
        f32_to_f16_k<<<(n + 255) / 256, 256, 0, stream>>>(s, d, n);
    };
    for (int i = 0; i < 7; ++i) conv(augWs[i], whAug[i], augOut[i] * augIn[i]);
    for (int i = 0; i < 5; ++i) conv(odeWs[i], whOde[i], odeOut[i] * odeIn[i]);
    for (int i = 0; i < 7; ++i) conv(decWs[i], whDec[i], decOut[i] * decIn[i]);

    // ---- GEMM dispatch: big tiles for big-M, small tiles for small-M ----
    auto gemm = [&](const float* A, const _Float16* Wh, const float* bias, float* Cout,
                    int M, int N, int K, const float* alpha_p, bool prelu) {
        if (M >= 512 && (M % 64) == 0 && (N % 128) == 0) {
            dim3 grid(N / 128, M / 64);
            if (prelu)
                gemm_tile<1, 64, 128, true><<<grid, 256, 0, stream>>>(A, Wh, bias, Cout, M, N, K, alpha_p);
            else
                gemm_tile<0, 64, 128, true><<<grid, 256, 0, stream>>>(A, Wh, bias, Cout, M, N, K, nullptr);
        } else {
            dim3 grid((N + 63) / 64, M / 32);
            if (prelu)
                gemm_tile<1, 32, 64, true><<<grid, 256, 0, stream>>>(A, Wh, bias, Cout, M, N, K, alpha_p);
            else if ((N % 64) == 0)
                gemm_tile<0, 32, 64, true><<<grid, 256, 0, stream>>>(A, Wh, bias, Cout, M, N, K, nullptr);
            else
                gemm_tile<0, 32, 64, false><<<grid, 256, 0, stream>>>(A, Wh, bias, Cout, M, N, K, nullptr);
        }
    };

    // ---- aug_net: Linear->PReLU x6, final Linear ----
    gemm(x, whAug[0], augBs[0], actA, B, U, IN, augAl[0], true);
    for (int i = 1; i < 6; ++i) {
        const float* in = (i & 1) ? actA : actB;
        float*      out = (i & 1) ? actB : actA;
        gemm(in, whAug[i], augBs[i], out, B, U, U, augAl[i], true);
    }
    gemm(actB, whAug[6], augBs[6], hbuf, B, L, U, nullptr, false);

    // ---- BatchNorm (training mode, batch stats) ----
    batchnorm_k<<<L, B, 0, stream>>>(hbuf, bn_g, bn_b, y, B, L);
    scatter_sol_k<<<(B * L + 255) / 256, 256, 0, stream>>>(y, sol, 0, B, T, L);

    // ---- ODE field: Linear->LayerNorm->ELU x4, final Linear ----
    auto odefunc = [&](const float* yin, float* kout) {
        gemm(yin, whOde[0], odeBs[0], actA, B, U, L, nullptr, false);
        layernorm_elu_k<<<B, 256, 0, stream>>>(actA, odeGs[0], odeBes[0], actB, U);
        for (int i = 1; i < 4; ++i) {
            gemm(actB, whOde[i], odeBs[i], actA, B, U, U, nullptr, false);
            layernorm_elu_k<<<B, 256, 0, stream>>>(actA, odeGs[i], odeBes[i], actB, U);
        }
        gemm(actB, whOde[4], odeBs[4], kout, B, L, U, nullptr, false);
    };

    // ---- RK4, one step per output interval (dt = 1/63) ----
    const float DT = 1.0f / 63.0f;
    const int n = B * L, eb = (n + 255) / 256;
    for (int step = 1; step < T; ++step) {
        odefunc(y, k1);
        axpy_k<<<eb, 256, 0, stream>>>(y, k1, 0.5f * DT, ytmp, n);
        odefunc(ytmp, k2);
        axpy_k<<<eb, 256, 0, stream>>>(y, k2, 0.5f * DT, ytmp, n);
        odefunc(ytmp, k3);
        axpy_k<<<eb, 256, 0, stream>>>(y, k3, DT, ytmp, n);
        odefunc(ytmp, k4);
        rk4_k<<<eb, 256, 0, stream>>>(y, k1, k2, k3, k4, DT, n);
        scatter_sol_k<<<eb, 256, 0, stream>>>(y, sol, step, B, T, L);
    }

    // ---- decoder over [B*T, L] in chunks of 2048 rows ----
    const int NCH = (B * T) / CH;  // 8
    for (int c = 0; c < NCH; ++c) {
        const float* in0 = sol + (size_t)c * CH * L;
        gemm(in0, whDec[0], decBs[0], dbufA, CH, U, L, decAl[0], true);
        for (int i = 1; i < 6; ++i) {
            const float* in = (i & 1) ? dbufA : dbufB;
            float*      out = (i & 1) ? dbufB : dbufA;
            gemm(in, whDec[i], decBs[i], out, CH, U, U, decAl[i], true);
        }
        gemm(dbufB, whDec[6], decBs[6], (float*)d_out + (size_t)c * CH, CH, 1, U,
             nullptr, false);
    }
}
